// GATLayer_68951404970605
// MI455X (gfx1250) — matrix operand order repaired and checked by reading, not compiled, not run on hardware
//
#include <hip/hip_runtime.h>

// GATConv forward for MI455X (gfx1250, wave32).
// [W->bf16 transpose] -> [WMMA bf16 GEMM h=xW, async-to-LDS double buffered]
// -> [logits] -> [init] -> [edge lrelu + atomic segmax] -> [exp + segsum]
// -> [fused scatter-aggregate with head-mean folded into out].

#define IN_CH 256
#define HEADS 4
#define CPH   64
#define HC    256           // HEADS*CPH
#define SLOPE 0.2f

typedef __attribute__((ext_vector_type(8)))  __bf16 v8bf;
typedef __attribute__((ext_vector_type(16))) __bf16 v16bf;
typedef __attribute__((ext_vector_type(8)))  float  v8f;
typedef __attribute__((__vector_size__(4 * sizeof(int)))) int i32x4;

#define XSF_STRIDE 36   // f32 x-tile rows: 144B stride, 16B-aligned chunks
#define WT_STRIDE  40   // bf16 W-tile rows: 80B stride, 16B-aligned chunks

// ---- CDNA5 async global->LDS path (guarded; sync fallback otherwise) ------
#if defined(__has_builtin)
#  if __has_builtin(__builtin_amdgcn_global_load_async_to_lds_b128) && \
      __has_builtin(__builtin_amdgcn_s_wait_asynccnt)
#    define USE_ASYNC 1
#  endif
#endif
#ifndef USE_ASYNC
#  define USE_ASYNC 0
#endif

#define GAS __attribute__((address_space(1)))
#define LAS __attribute__((address_space(3)))

#if USE_ASYNC
#  define WAIT_ASYNC(n)       __builtin_amdgcn_s_wait_asynccnt(n)
#  define ASYNC_CP_B128(g, l) __builtin_amdgcn_global_load_async_to_lds_b128( \
                                  (GAS i32x4*)(g), (LAS i32x4*)(l), 0, 0)
#else
#  define WAIT_ASYNC(n)
#endif

// ---------------------------------------------------------------------------
// Kernel 0: Wt[n][k] = bf16(W[k][n])  (one-time transpose+convert, 128 KB out)
// ---------------------------------------------------------------------------
__global__ void gat_wconv(const float* __restrict__ W, __bf16* __restrict__ Wt)
{
  const int idx = blockIdx.x * blockDim.x + threadIdx.x;   // k*256 + n
  if (idx >= IN_CH * HC) return;
  const int k = idx >> 8, n = idx & 255;
  Wt[(size_t)n * IN_CH + k] = (__bf16)W[idx];
}

// ---------------------------------------------------------------------------
// Kernel 1: h = x @ W  (M=Nn, K=256, N=256), bf16 WMMA, fp32 accumulate.
// Block: 256 thr = 8 waves; tile 64(M) x 256(N); double-buffered LDS with
// async global->LDS copies overlapping the WMMA inner loop.
// ---------------------------------------------------------------------------
__device__ __forceinline__ void stage_tiles(
    const float* __restrict__ x, const __bf16* __restrict__ Wt,
    float* __restrict__ xsb, __bf16* __restrict__ wtb,
    int m0, int kb, int Nn, int t)
{
  // x tile: 64 rows x 32 k (f32) = 512 chunks of 16B -> 2 per thread
  #pragma unroll
  for (int i = 0; i < 2; ++i) {
    const int cid = t + 256 * i;
    const int row = cid >> 3, g = cid & 7;
    int grow = m0 + row; if (grow >= Nn) grow = Nn - 1;    // clamp, no diverge
    const float* gp = x + (size_t)grow * IN_CH + kb * 32 + g * 4;
    float*       lp = xsb + row * XSF_STRIDE + g * 4;
#if USE_ASYNC
    ASYNC_CP_B128(gp, lp);
#else
    *(float4*)lp = *(const float4*)gp;
#endif
  }
  // W tile: 256 n x 32 k (bf16) = 1024 chunks of 16B -> 4 per thread
  #pragma unroll
  for (int i = 0; i < 4; ++i) {
    const int cid = t + 256 * i;
    const int n = cid >> 2, g = cid & 3;
    const __bf16* gp = Wt + (size_t)n * IN_CH + kb * 32 + g * 8;
    __bf16*       lp = wtb + n * WT_STRIDE + g * 8;
#if USE_ASYNC
    ASYNC_CP_B128(gp, lp);
#else
    *(uint4*)(void*)lp = *(const uint4*)(const void*)gp;
#endif
  }
}

__global__ __launch_bounds__(256) void gat_gemm_wmma(
    const float* __restrict__ x, const __bf16* __restrict__ Wt,
    float* __restrict__ h, int Nn)
{
  __shared__ float  xsf[2][64 * XSF_STRIDE];    // x tile (f32, raw DMA)
  __shared__ __bf16 wtb[2][256 * WT_STRIDE];    // W tile transposed (bf16)

  const int t     = threadIdx.x;
  const int wave  = t >> 5;
  const int lane  = t & 31;
  const int l15   = lane & 15;
  const int lhi   = lane >> 4;      // 0: lanes 0-15, 1: lanes 16-31
  const int msub  = wave & 3;
  const int nhalf = wave >> 2;
  const int m0    = blockIdx.x * 64;

  v8f zf = {0.f,0.f,0.f,0.f,0.f,0.f,0.f,0.f};
  v8f acc[8];
  #pragma unroll
  for (int i = 0; i < 8; ++i) acc[i] = zf;

  stage_tiles(x, Wt, xsf[0], wtb[0], m0, 0, Nn, t);   // preload kb=0
  int cur = 0;

  for (int kb = 0; kb < 8; ++kb) {            // K = 256 in steps of 32
    if (kb < 7) {                             // prefetch next k-step (overlap)
      stage_tiles(x, Wt, xsf[cur ^ 1], wtb[cur ^ 1], m0, kb + 1, Nn, t);
      WAIT_ASYNC(6);                          // 6 newest pending = next tile
    } else {
      WAIT_ASYNC(0);
    }
    __syncthreads();                          // all waves' copies visible

    const float*  xsc = xsf[cur];
    const __bf16* wtc = wtb[cur];

    // A fragment (16x32 bf16, ISA layout): lanes<16 K=0..7,16..23;
    // lanes>=16 K=8..15,24..31. Convert f32->bf16 here.
    const int am  = msub * 16 + l15;
    const int ac0 = lhi * 8;
    const float* ap = xsc + am * XSF_STRIDE + ac0;
    const float4 f0 = *(const float4*)(ap);
    const float4 f1 = *(const float4*)(ap + 4);
    const float4 f2 = *(const float4*)(ap + 16);
    const float4 f3 = *(const float4*)(ap + 20);
    v16bf afrag;
    afrag[0]=(__bf16)f0.x; afrag[1]=(__bf16)f0.y; afrag[2]=(__bf16)f0.z; afrag[3]=(__bf16)f0.w;
    afrag[4]=(__bf16)f1.x; afrag[5]=(__bf16)f1.y; afrag[6]=(__bf16)f1.z; afrag[7]=(__bf16)f1.w;
    afrag[8]=(__bf16)f2.x; afrag[9]=(__bf16)f2.y; afrag[10]=(__bf16)f2.z; afrag[11]=(__bf16)f2.w;
    afrag[12]=(__bf16)f3.x; afrag[13]=(__bf16)f3.y; afrag[14]=(__bf16)f3.z; afrag[15]=(__bf16)f3.w;

    #pragma unroll
    for (int nt = 0; nt < 8; ++nt) {
      // B fragment (32x16 bf16, ISA layout): N=l15; lanes<16 K=0..7,16..23;
      // lanes>=16 K=8..15,24..31; from transposed wt rows.
      const int bn  = nhalf * 128 + nt * 16 + l15;
      const int bk0 = lhi * 8;
      const v8bf blo = *(const v8bf*)&wtc[bn * WT_STRIDE + bk0];
      const v8bf bhi = *(const v8bf*)&wtc[bn * WT_STRIDE + bk0 + 16];
      const v16bf bfrag = __builtin_shufflevector(blo, bhi,
          0,1,2,3,4,5,6,7,8,9,10,11,12,13,14,15);
      acc[nt] = __builtin_amdgcn_wmma_f32_16x16x32_bf16(
          false, afrag, false, bfrag, (short)0, acc[nt], false, false);
    }
    __syncthreads();                          // compute done before reuse
    cur ^= 1;
  }

  // Store C: VGPR r -> M = r + 8*lhi, N = l15. Strip validity is uniform
  // per 16-row m-subtile -> hoist the guard (fast path: no predication).
  const int rowbase = m0 + msub * 16 + lhi * 8;
  if (m0 + msub * 16 + 16 <= Nn) {
    #pragma unroll
    for (int nt = 0; nt < 8; ++nt) {
      const int ncol = nhalf * 128 + nt * 16 + l15;
      #pragma unroll
      for (int r = 0; r < 8; ++r)
        h[(size_t)(rowbase + r) * HC + ncol] = acc[nt][r];
    }
  } else {
    #pragma unroll
    for (int nt = 0; nt < 8; ++nt) {
      const int ncol = nhalf * 128 + nt * 16 + l15;
      #pragma unroll
      for (int r = 0; r < 8; ++r)
        if (rowbase + r < Nn) h[(size_t)(rowbase + r) * HC + ncol] = acc[nt][r];
    }
  }
}

// ---------------------------------------------------------------------------
// Kernel 2: per-(node,head) attention logits (dots of length 64).
// ---------------------------------------------------------------------------
__global__ void gat_logits(const float* __restrict__ h,
                           const float* __restrict__ att_src,
                           const float* __restrict__ att_dst,
                           float* __restrict__ a_src,
                           float* __restrict__ a_dst, int Nn)
{
  const int idx = blockIdx.x * blockDim.x + threadIdx.x;
  if (idx >= Nn * HEADS) return;
  const int n = idx >> 2, hh = idx & 3;
  const float4* hp = (const float4*)(h + (size_t)n * HC + hh * CPH);
  const float4* sp = (const float4*)(att_src + hh * CPH);
  const float4* dp = (const float4*)(att_dst + hh * CPH);
  float ss = 0.f, sd = 0.f;
  #pragma unroll
  for (int i = 0; i < 16; ++i) {
    const float4 hv = hp[i], av = sp[i], bv = dp[i];
    ss += hv.x*av.x + hv.y*av.y + hv.z*av.z + hv.w*av.w;
    sd += hv.x*bv.x + hv.y*bv.y + hv.z*bv.z + hv.w*bv.w;
  }
  a_src[idx] = ss;
  a_dst[idx] = sd;
}

// ---------------------------------------------------------------------------
// Kernel 3: init out = bias (head-mean accumulator), amax = -inf, denom = 0.
// ---------------------------------------------------------------------------
__global__ void gat_init(float* __restrict__ out, const float* __restrict__ bias,
                         float* __restrict__ amax, float* __restrict__ denom,
                         int Nn)
{
  const int idx = blockIdx.x * blockDim.x + threadIdx.x;
  if (idx < Nn * CPH)   out[idx] = bias[idx & (CPH - 1)];
  if (idx < Nn * HEADS) { amax[idx] = -3.0e38f; denom[idx] = 0.f; }
}

__device__ __forceinline__ void atomicMaxF(float* addr, float v) {
  if (v >= 0.f) atomicMax((int*)addr, __float_as_int(v));
  else          atomicMin((unsigned int*)addr, (unsigned int)__float_as_int(v));
}

// ---------------------------------------------------------------------------
// Kernel 4: alpha = leaky_relu(a_src[s]+a_dst[d]); atomic segment max.
// Edges e<E from edge_index; e>=E are self loops (s=d=e-E).
// ---------------------------------------------------------------------------
__global__ void gat_alpha_max(const int* __restrict__ ei, int E, int Nn,
                              const float* __restrict__ a_src,
                              const float* __restrict__ a_dst,
                              float* __restrict__ alpha,
                              float* __restrict__ amax)
{
  const int e = blockIdx.x * blockDim.x + threadIdx.x;
  if (e >= E + Nn) return;
  int s, d;
  if (e < E) { s = ei[e]; d = ei[E + e]; } else { s = d = e - E; }
  const float4 as = *(const float4*)(a_src + (size_t)s * 4);
  const float4 ad = *(const float4*)(a_dst + (size_t)d * 4);
  float4 al;
  al.x = as.x + ad.x; al.x = al.x > 0.f ? al.x : SLOPE * al.x;
  al.y = as.y + ad.y; al.y = al.y > 0.f ? al.y : SLOPE * al.y;
  al.z = as.z + ad.z; al.z = al.z > 0.f ? al.z : SLOPE * al.z;
  al.w = as.w + ad.w; al.w = al.w > 0.f ? al.w : SLOPE * al.w;
  *(float4*)(alpha + (size_t)e * 4) = al;
  atomicMaxF(&amax[(size_t)d * 4 + 0], al.x);
  atomicMaxF(&amax[(size_t)d * 4 + 1], al.y);
  atomicMaxF(&amax[(size_t)d * 4 + 2], al.z);
  atomicMaxF(&amax[(size_t)d * 4 + 3], al.w);
}

// ---------------------------------------------------------------------------
// Kernel 5: ex = exp(alpha - amax[d]) (in place); atomic denom sum.
// ---------------------------------------------------------------------------
__global__ void gat_exp_sum(const int* __restrict__ ei, int E, int Nn,
                            const float* __restrict__ amax,
                            float* __restrict__ ex,       // in: alpha, out: ex
                            float* __restrict__ denom)
{
  const int e = blockIdx.x * blockDim.x + threadIdx.x;
  if (e >= E + Nn) return;
  const int d = (e < E) ? ei[E + e] : (e - E);
  const float4 al = *(const float4*)(ex + (size_t)e * 4);
  const float4 mx = *(const float4*)(amax + (size_t)d * 4);
  float4 ev;
  ev.x = __expf(al.x - mx.x);
  ev.y = __expf(al.y - mx.y);
  ev.z = __expf(al.z - mx.z);
  ev.w = __expf(al.w - mx.w);
  *(float4*)(ex + (size_t)e * 4) = ev;
  atomicAdd(&denom[(size_t)d * 4 + 0], ev.x);
  atomicAdd(&denom[(size_t)d * 4 + 1], ev.y);
  atomicAdd(&denom[(size_t)d * 4 + 2], ev.z);
  atomicAdd(&denom[(size_t)d * 4 + 3], ev.w);
}

// ---------------------------------------------------------------------------
// Kernel 6: out[d,c] += 0.25 * sum_h coef[e,h] * h[s, h*64 + c].
// Block 256 thr = 4 edges x 64 channels; 1 atomicAdd per (edge,channel).
// ---------------------------------------------------------------------------
__global__ __launch_bounds__(256) void gat_aggregate(
    const int* __restrict__ ei, int E, int Nn,
    const float* __restrict__ h, const float* __restrict__ ex,
    const float* __restrict__ denom, float* __restrict__ out)
{
  const int slot = threadIdx.x >> 6;
  const int c    = threadIdx.x & 63;
  const long long e = (long long)blockIdx.x * 4 + slot;
  if (e >= (long long)E + Nn) return;
  int s, d;
  if (e < E) { s = ei[e]; d = ei[E + (int)e]; } else { s = d = (int)e - E; }
  float accv = 0.f;
  #pragma unroll
  for (int hh = 0; hh < HEADS; ++hh) {
    const float coef = ex[(size_t)e * 4 + hh] /
                       (denom[(size_t)d * 4 + hh] + 1e-16f);
    accv += coef * h[(size_t)s * HC + hh * CPH + c];
  }
  atomicAdd(&out[(size_t)d * CPH + c], 0.25f * accv);
}

// ---------------------------------------------------------------------------
extern "C" void kernel_launch(void* const* d_in, const int* in_sizes, int n_in,
                              void* d_out, int out_size, void* d_ws, size_t ws_size,
                              hipStream_t stream)
{
  const float* x       = (const float*)d_in[0];
  const int*   ei      = (const int*)  d_in[1];   // [2, E]: row0=src, row1=dst
  const float* W       = (const float*)d_in[2];
  const float* att_src = (const float*)d_in[3];
  const float* att_dst = (const float*)d_in[4];
  const float* bias    = (const float*)d_in[5];
  float*       out     = (float*)d_out;           // [Nn, 64]

  const int Nn   = in_sizes[0] / IN_CH;
  const int E    = in_sizes[1] / 2;
  const int Etot = E + Nn;

  // Workspace (floats): h | a_src | a_dst | amax | denom | ex | Wt(bf16)
  float* h_buf = (float*)d_ws;
  float* a_src = h_buf + (size_t)Nn * HC;
  float* a_dst = a_src + (size_t)Nn * HEADS;
  float* amax  = a_dst + (size_t)Nn * HEADS;
  float* denom = amax  + (size_t)Nn * HEADS;
  float* exbuf = denom + (size_t)Nn * HEADS;        // [Etot, 4]
  __bf16* Wt   = (__bf16*)(exbuf + (size_t)Etot * HEADS);   // [256 n][256 k]

  // 0. W -> bf16 transposed
  gat_wconv<<<(IN_CH * HC + 255) / 256, 256, 0, stream>>>(W, Wt);
  // 1. GEMM: h = x @ W (WMMA bf16, async double-buffered LDS)
  gat_gemm_wmma<<<(Nn + 63) / 64, 256, 0, stream>>>(x, Wt, h_buf, Nn);
  // 2. Attention logits
  gat_logits<<<(Nn * HEADS + 255) / 256, 256, 0, stream>>>(
      h_buf, att_src, att_dst, a_src, a_dst, Nn);
  // 3. Init accumulators
  gat_init<<<(Nn * CPH + 255) / 256, 256, 0, stream>>>(out, bias, amax, denom, Nn);
  // 4. Edge alpha + segment max
  gat_alpha_max<<<(Etot + 255) / 256, 256, 0, stream>>>(
      ei, E, Nn, a_src, a_dst, exbuf, amax);
  // 5. Edge exp + segment sum
  gat_exp_sum<<<(Etot + 255) / 256, 256, 0, stream>>>(
      ei, E, Nn, amax, exbuf, denom);
  // 6. Scatter-aggregate (head-mean fused, bias already in out)
  gat_aggregate<<<(Etot + 3) / 4, 256, 0, stream>>>(
      ei, E, Nn, h_buf, exbuf, denom, out);
}
